// StackSAModuleMSG_51015621542422
// MI455X (gfx1250) — matrix operand order repaired and checked by reading, not compiled, not run on hardware
//
#include <hip/hip_runtime.h>
#include <hip/hip_bf16.h>

typedef __attribute__((ext_vector_type(16))) _Float16 v16h;
typedef __attribute__((ext_vector_type(8)))  float    v8f;

#define NBATCH   2
#define NB       16384
#define MB       2048
#define M_TOTAL  (NBATCH * MB)   // 4096
#define K0REAL   67
#define K0PAD    96
#define WHALVES  18432           // packed weight halves per scale (L0:6144 L1:4096 L2:8192)

// ---- workspace layout (bytes) ----
#define OFF_W    0                         // 2*18432 halves  = 73728 B
#define OFF_BN   73728                     // 1024 floats     = 4096 B (scale[512] then bias[512])
#define OFF_CNT  77824                     // 8192 ints       = 32768 B
#define OFF_A0   110592                    // 4096*16*96 f16  = 12582912 B
#define OFF_A1   12693504                  // 4096*32*96 f16  = 25165824 B  (end 37859328)

struct MlpParams {
    const float* W[2][3];
    const float* gamma[2][3];
    const float* beta[2][3];
    const float* mu[2][3];
    const float* var[2][3];
};

// ------------------------------------------------------------------
// Fold BN into (scale, bias); repack weights into WMMA B-operand
// lane layout: halfidx -> v = h/2, half = h&1;
//   K = kc*32 + (v<4?0:16) + (lane>>4)*8 + 2(v&3) + half,  N = ntile*16 + (lane&15)
// one v16h per lane stored contiguously => single 32B load in MLP.
// ------------------------------------------------------------------
__global__ void pack_kernel(MlpParams p, _Float16* __restrict__ wpack,
                            float* __restrict__ bnscale, float* __restrict__ bnbias) {
    int i = blockIdx.x * blockDim.x + threadIdx.x;
    if (i < 2 * WHALVES) {
        int s = i / WHALVES, r = i % WHALVES;
        int base, ntl, Kin;
        if (r < 6144)       { base = 0;     ntl = 4; Kin = K0REAL; }
        else if (r < 10240) { base = 6144;  ntl = 4; Kin = 64;     }
        else                { base = 10240; ntl = 8; Kin = 64;     }
        int layer = (r < 6144) ? 0 : (r < 10240 ? 1 : 2);
        int e     = r - base;
        int tile  = e >> 9;            // 512 halves per (kc,ntile) tile
        int kc    = tile / ntl;
        int nt    = tile % ntl;
        int lane  = (e >> 4) & 31;
        int h     = e & 15;
        int v     = h >> 1, half = h & 1;
        int K     = kc * 32 + ((v < 4) ? 0 : 16) + (lane >> 4) * 8 + 2 * (v & 3) + half;
        int N     = nt * 16 + (lane & 15);
        float w   = (K < Kin) ? p.W[s][layer][N * Kin + K] : 0.0f;
        wpack[i]  = (_Float16)w;
    } else {
        int j = i - 2 * WHALVES;
        if (j < 512) {
            int s = j >> 8, c = j & 255;
            int layer = (c < 64) ? 0 : (c < 128 ? 1 : 2);
            int cl    = c - ((layer == 0) ? 0 : (layer == 1) ? 64 : 128);
            float sc  = p.gamma[s][layer][cl] * rsqrtf(p.var[s][layer][cl] + 1e-5f);
            bnscale[s * 256 + c] = sc;
            bnbias [s * 256 + c] = p.beta[s][layer][cl] - p.mu[s][layer][cl] * sc;
        }
    }
}

// ------------------------------------------------------------------
// Ball query (first nsample in-range, in index order, via ballot
// prefix sums) + stage f16 A rows [relxyz | feat | zero-pad] (96 cols)
// One wave per (keypoint, scale).
// ------------------------------------------------------------------
__global__ __launch_bounds__(32)
void bq_kernel(const float* __restrict__ xyz, const float* __restrict__ new_xyz,
               const float* __restrict__ feat,
               _Float16* __restrict__ A0, _Float16* __restrict__ A1,
               int* __restrict__ cnts) {
    const int m    = blockIdx.x;
    const int s    = blockIdx.y;
    const int lane = threadIdx.x;
    const float r2 = (s == 0) ? 0.64f : 2.56f;
    const int  ns  = (s == 0) ? 16 : 32;
    _Float16*  A   = (s == 0) ? A0 : A1;

    const int   b     = m / MB;
    const int   base0 = b * NB;
    const float nx = new_xyz[m * 3 + 0];
    const float ny = new_xyz[m * 3 + 1];
    const float nz = new_xyz[m * 3 + 2];

    __shared__ int idxbuf[32];
    int cnt = 0;
    for (int base = 0; base < NB && cnt < ns; base += 32) {
        int j = base0 + base + lane;
        float dx = xyz[j * 3 + 0] - nx;
        float dy = xyz[j * 3 + 1] - ny;
        float dz = xyz[j * 3 + 2] - nz;
        bool in = (dx * dx + dy * dy + dz * dz) < r2;
        unsigned mask = (unsigned)__ballot(in);
        int pre = __popc(mask & ((1u << lane) - 1u));
        if (in && (cnt + pre) < ns) idxbuf[cnt + pre] = j;
        cnt += __popc(mask);
    }
    if (cnt > ns) cnt = ns;
    if (lane == 0) cnts[s * M_TOTAL + m] = cnt;
    __syncthreads();

    _Float16* Abase = A + (size_t)m * ns * K0PAD;
    for (int t = 0; t < ns; ++t) {
        bool valid = t < cnt;
        int  j     = valid ? idxbuf[t] : base0;
#pragma unroll
        for (int cc = 0; cc < 3; ++cc) {
            int   c = lane + cc * 32;
            float v = 0.0f;
            if (valid) {
                if (c < 3)           v = xyz[j * 3 + c] - new_xyz[m * 3 + c];
                else if (c < K0REAL) v = feat[(size_t)j * 64 + (c - 3)];
            }
            Abase[(size_t)t * K0PAD + c] = (_Float16)v;
        }
    }
}

// A-operand fragment from row-major f16 tile (ISA 16x32 f16 A layout):
// lane l: M = l&15;  K(v) = (v<4?0:16) + (l>>4)*8 + 2*(v&3)
#define LOAD_A_FRAG(dst, src_rowbase, stride)                                   \
    {                                                                           \
        _Pragma("unroll") for (int v_ = 0; v_ < 8; ++v_) {                      \
            int k_ = ((v_ < 4) ? 0 : 16) + grp * 8 + 2 * (v_ & 3);              \
            dst[2 * v_]     = (src_rowbase)[row * (stride) + k_];               \
            dst[2 * v_ + 1] = (src_rowbase)[row * (stride) + k_ + 1];           \
        }                                                                       \
    }

// ------------------------------------------------------------------
// Fused 3-layer MLP + BN + ReLU + masked max-pool. 8 waves / block,
// one keypoint per wave, NS/16 sample-row tiles per keypoint.
// ------------------------------------------------------------------
template <int NS>
__global__ __launch_bounds__(256)
void mlp_kernel(const _Float16* __restrict__ Astage, const _Float16* __restrict__ wpack,
                const float* __restrict__ bnscale, const float* __restrict__ bnbias,
                const int* __restrict__ cnts, float* __restrict__ out, int outoff) {
    const int wave = threadIdx.x >> 5;
    const int lane = threadIdx.x & 31;
    const int grp  = lane >> 4;
    const int row  = lane & 15;
    const int m    = blockIdx.x * 8 + wave;

    __shared__ _Float16 ldsH[8][16 * 64];
    const v16h* bfrag = (const v16h*)wpack;   // contiguous 32B per lane per tile
    const int   cnt   = cnts[m];

    float colmax[8];
#pragma unroll
    for (int i = 0; i < 8; ++i) colmax[i] = 0.0f;

    constexpr int NT = NS / 16;
    for (int tile = 0; tile < NT; ++tile) {
        // ---- layer 0: A from global staging, K = 96 (3 chunks) ----
        const _Float16* tb = Astage + ((size_t)m * NS + tile * 16) * K0PAD;
        v16h a0[3];
#pragma unroll
        for (int kc = 0; kc < 3; ++kc) LOAD_A_FRAG(a0[kc], tb + kc * 32, K0PAD);

#pragma unroll
        for (int nt = 0; nt < 4; ++nt) {
            v8f acc = {};
#pragma unroll
            for (int kc = 0; kc < 3; ++kc) {
                v16h bt = bfrag[(kc * 4 + nt) * 32 + lane];
                acc = __builtin_amdgcn_wmma_f32_16x16x32_f16(false, a0[kc], false, bt,
                                                             (short)0, acc, false, false);
            }
            int   col = nt * 16 + row;
            float sc = bnscale[col], bi = bnbias[col];
#pragma unroll
            for (int r = 0; r < 8; ++r)
                ldsH[wave][(r + 8 * grp) * 64 + col] = (_Float16)fmaxf(acc[r] * sc + bi, 0.0f);
        }
        __syncthreads();

        // ---- layer 1: A from LDS, K = 64 (2 chunks) ----
        v16h a1[2];
#pragma unroll
        for (int kc = 0; kc < 2; ++kc) LOAD_A_FRAG(a1[kc], &ldsH[wave][kc * 32], 64);
        __syncthreads();

#pragma unroll
        for (int nt = 0; nt < 4; ++nt) {
            v8f acc = {};
#pragma unroll
            for (int kc = 0; kc < 2; ++kc) {
                v16h bt = bfrag[384 + (kc * 4 + nt) * 32 + lane];   // L1 base: 6144/16
                acc = __builtin_amdgcn_wmma_f32_16x16x32_f16(false, a1[kc], false, bt,
                                                             (short)0, acc, false, false);
            }
            int   col = nt * 16 + row;
            float sc = bnscale[64 + col], bi = bnbias[64 + col];
#pragma unroll
            for (int r = 0; r < 8; ++r)
                ldsH[wave][(r + 8 * grp) * 64 + col] = (_Float16)fmaxf(acc[r] * sc + bi, 0.0f);
        }
        __syncthreads();

        // ---- layer 2: K = 64, N = 128 (8 tiles); epilogue masks + max ----
        v16h a2[2];
#pragma unroll
        for (int kc = 0; kc < 2; ++kc) LOAD_A_FRAG(a2[kc], &ldsH[wave][kc * 32], 64);

#pragma unroll
        for (int nt = 0; nt < 8; ++nt) {
            v8f acc = {};
#pragma unroll
            for (int kc = 0; kc < 2; ++kc) {
                v16h bt = bfrag[640 + (kc * 8 + nt) * 32 + lane];   // L2 base: 10240/16
                acc = __builtin_amdgcn_wmma_f32_16x16x32_f16(false, a2[kc], false, bt,
                                                             (short)0, acc, false, false);
            }
            int   col = nt * 16 + row;
            float sc = bnscale[128 + col], bi = bnbias[128 + col];
            float mx = colmax[nt];
#pragma unroll
            for (int r = 0; r < 8; ++r) {
                int   samp = tile * 16 + r + 8 * grp;   // D row for this lane/VGPR
                float h    = fmaxf(acc[r] * sc + bi, 0.0f);
                if (samp < cnt) mx = fmaxf(mx, h);
            }
            colmax[nt] = mx;
        }
        __syncthreads();
    }

    // cross-half column reduce (D layout: rows split across lane halves)
#pragma unroll
    for (int nt = 0; nt < 8; ++nt) {
        float mx = fmaxf(colmax[nt], __shfl_xor(colmax[nt], 16, 32));
        if (grp == 0) out[(size_t)m * 256 + outoff + nt * 16 + row] = mx;
    }
}

extern "C" void kernel_launch(void* const* d_in, const int* in_sizes, int n_in,
                              void* d_out, int out_size, void* d_ws, size_t ws_size,
                              hipStream_t stream) {
    const float* xyz     = (const float*)d_in[0];
    const float* new_xyz = (const float*)d_in[2];
    const float* feat    = (const float*)d_in[4];

    MlpParams P;
    for (int s = 0; s < 2; ++s)
        for (int l = 0; l < 3; ++l) {
            int base       = 5 + s * 15 + l * 5;
            P.W[s][l]      = (const float*)d_in[base + 0];
            P.gamma[s][l]  = (const float*)d_in[base + 1];
            P.beta[s][l]   = (const float*)d_in[base + 2];
            P.mu[s][l]     = (const float*)d_in[base + 3];
            P.var[s][l]    = (const float*)d_in[base + 4];
        }

    char*      ws      = (char*)d_ws;
    _Float16*  wpack   = (_Float16*)(ws + OFF_W);
    float*     bnscale = (float*)(ws + OFF_BN);
    float*     bnbias  = bnscale + 512;
    int*       cnts    = (int*)(ws + OFF_CNT);
    _Float16*  A0      = (_Float16*)(ws + OFF_A0);
    _Float16*  A1      = (_Float16*)(ws + OFF_A1);
    float*     out     = (float*)d_out;

    pack_kernel<<<(2 * WHALVES + 512 + 255) / 256, 256, 0, stream>>>(P, wpack, bnscale, bnbias);

    dim3 gbq(M_TOTAL, 2);
    bq_kernel<<<gbq, 32, 0, stream>>>(xyz, new_xyz, feat, A0, A1, cnts);

    mlp_kernel<16><<<M_TOTAL / 8, 256, 0, stream>>>(A0, wpack, bnscale, bnbias,
                                                    cnts, out, 0);
    mlp_kernel<32><<<M_TOTAL / 8, 256, 0, stream>>>(A1, wpack + WHALVES, bnscale + 256,
                                                    bnbias + 256, cnts + M_TOTAL, out, 128);
}
